// Autoformer_43980465111384
// MI455X (gfx1250) — compile-verified, matrix operand order checked by reading, not run on hardware
//
#include <hip/hip_runtime.h>
#include <math.h>

// ---------------------------------------------------------------------------
// Autoformer forward for gfx1250 (CDNA5, wave32, WMMA).
// GEMMs: LDS-staged bf16 tiles, v_wmma_f32_16x16x32_bf16, f32 accumulate.
// Autocorrelation done exactly via batched QK^T GEMM + wrapped-diagonal
// reduction (channel-mean commutes with the FFT correlation).
// ---------------------------------------------------------------------------

typedef __attribute__((ext_vector_type(16))) __bf16 v16bf;
typedef __attribute__((ext_vector_type(8)))  float  v8f;

#define NB    32
#define LL    512
#define DDIM  512
#define DFF   2048
#define CINC  7
#define DMK   4
#define NTOP  6
#define PREDN 256

static constexpr long BL  = (long)NB * LL;          // 16384 token rows
static constexpr long BLD = (long)NB * LL * DDIM;   // 8,388,608 floats

__device__ __forceinline__ unsigned pack_bf16(float x, float y) {
  union { __bf16 h[2]; unsigned u; } p;
  p.h[0] = (__bf16)x;
  p.h[1] = (__bf16)y;
  return p.u;
}

union FragU { unsigned u[8]; v16bf v; };

// C[M,N] = act(A[M,K] @ W + bias) + R.
// W row-major [K,N] (TRANSB=0) or [N,K] (TRANSB=1).
// Block = 128 threads = 4 waves. Block tile 128(M) x 64(N), K-step 32.
// Wave w owns rows w*32..w*32+31: 2 A-frags x 4 B-frags = 8 WMMA / K-step.
template <int TRANSB, int ACT>
__global__ void __launch_bounds__(128)
gemm_wmma(const float* __restrict__ A, long sA,
          const float* __restrict__ W, long sW,
          const float* __restrict__ bias,
          const float* __restrict__ R, long sR,
          float* __restrict__ C, long sC,
          int M, int N, int K) {
  __shared__ unsigned lds_a[128 * 16];  // [row][k/2] bf16 pairs (8 KB)
  __shared__ unsigned lds_b[64 * 16];   // [n][k/2]  bf16 pairs (4 KB)

  int bz = blockIdx.z;
  A += (long)bz * sA;
  W += (long)bz * sW;
  C += (long)bz * sC;
  if (R) R += (long)bz * sR;

  int t    = threadIdx.x;
  int wave = t >> 5, lane = t & 31, half = lane >> 4, l16 = lane & 15;
  int mBase = blockIdx.y * 128;
  int nBase = blockIdx.x * 64;

  // Fragment dword offsets (k/2 units) for this lane (ISA 7.12.2 16-bit layout):
  // dword d<4 -> K = kb+2d ; d>=4 -> K = 16+kb+2(d-4), kb = (lane/16)*8.
  int h4 = half << 2;
  int kcol2[8];
#pragma unroll
  for (int d = 0; d < 8; ++d) kcol2[d] = (d < 4) ? (h4 + d) : (8 + h4 + (d - 4));

  v8f acc[2][4];
#pragma unroll
  for (int ms = 0; ms < 2; ++ms)
#pragma unroll
    for (int j = 0; j < 4; ++j) acc[ms][j] = v8f{};

  for (int k0 = 0; k0 < K; k0 += 32) {
    __syncthreads();
    // ---- stage A tile: 128x32 f32, coalesced float4, pack to bf16 ----
#pragma unroll
    for (int i = 0; i < 8; ++i) {
      int flat4 = t + 128 * i;            // 1024 float4 groups
      int row = flat4 >> 3;               // 8 groups per 32-wide row
      int c4  = (flat4 & 7) << 2;
      const float4 s = *reinterpret_cast<const float4*>(
          A + (long)(mBase + row) * K + k0 + c4);
      lds_a[row * 16 + (c4 >> 1)]     = pack_bf16(s.x, s.y);
      lds_a[row * 16 + (c4 >> 1) + 1] = pack_bf16(s.z, s.w);
    }
    // ---- stage B tile into [n][k] layout ----
    if (TRANSB) {
#pragma unroll
      for (int i = 0; i < 4; ++i) {
        int flat4 = t + 128 * i;          // 512 float4 groups (64x32)
        int n  = flat4 >> 3;
        int c4 = (flat4 & 7) << 2;
        const float4 s = *reinterpret_cast<const float4*>(
            W + (long)(nBase + n) * K + k0 + c4);
        lds_b[n * 16 + (c4 >> 1)]     = pack_bf16(s.x, s.y);
        lds_b[n * 16 + (c4 >> 1) + 1] = pack_bf16(s.z, s.w);
      }
    } else {
#pragma unroll
      for (int i = 0; i < 4; ++i) {
        int flat4 = t + 128 * i;          // 512 float4 groups (32x64)
        int k  = flat4 >> 4;              // 16 groups per 64-wide n row
        int n4 = (flat4 & 15) << 2;
        const float4 s = *reinterpret_cast<const float4*>(
            W + (long)(k0 + k) * N + nBase + n4);
        __bf16* lb = (__bf16*)lds_b;      // [n][k], k-stride 1, n-stride 32
        lb[(n4 + 0) * 32 + k] = (__bf16)s.x;
        lb[(n4 + 1) * 32 + k] = (__bf16)s.y;
        lb[(n4 + 2) * 32 + k] = (__bf16)s.z;
        lb[(n4 + 3) * 32 + k] = (__bf16)s.w;
      }
    }
    __syncthreads();

    // ---- build fragments from LDS and issue WMMAs ----
    FragU bfr[4];
#pragma unroll
    for (int j = 0; j < 4; ++j) {
      int nl = j * 16 + l16;
#pragma unroll
      for (int d = 0; d < 8; ++d) bfr[j].u[d] = lds_b[nl * 16 + kcol2[d]];
    }
#pragma unroll
    for (int ms = 0; ms < 2; ++ms) {
      int rowl = wave * 32 + ms * 16 + l16;
      FragU af;
#pragma unroll
      for (int d = 0; d < 8; ++d) af.u[d] = lds_a[rowl * 16 + kcol2[d]];
#pragma unroll
      for (int j = 0; j < 4; ++j)
        acc[ms][j] = __builtin_amdgcn_wmma_f32_16x16x32_bf16(
            false, af.v, false, bfr[j].v, (short)0, acc[ms][j], false, false);
    }
  }

  // ---- epilogue: bias, exact GELU, residual ----
#pragma unroll
  for (int ms = 0; ms < 2; ++ms)
#pragma unroll
    for (int j = 0; j < 4; ++j) {
      int n = nBase + j * 16 + l16;
#pragma unroll
      for (int i = 0; i < 8; ++i) {
        int m = mBase + wave * 32 + ms * 16 + i + 8 * half;
        float v = acc[ms][j][i];
        if (bias) v += bias[n];
        if (ACT) v = 0.5f * v * (1.0f + erff(v * 0.70710678118654752f));
        if (R) v += R[(long)m * N + n];
        C[(long)m * N + n] = v;
      }
    }
}

// mean_corr[b,tau] = (1/512) * sum_t G[b][t][(t-tau) mod L]
__global__ void mean_corr_kernel(const float* __restrict__ G, float* __restrict__ mc) {
  int b = blockIdx.x;
  const float* g = G + (long)b * LL * LL;
  for (int tau = threadIdx.x; tau < LL; tau += blockDim.x) {
    float s = 0.f;
    for (int t = 0; t < LL; ++t) s += g[(long)t * LL + ((t - tau) & (LL - 1))];
    mc[b * LL + tau] = s * (1.0f / 512.0f);
  }
}

// top-6 (descending, first-index ties) + softmax, one thread per batch
__global__ void topk_kernel(const float* __restrict__ mc,
                            float* __restrict__ wts, int* __restrict__ dly) {
  int b = threadIdx.x;
  if (b >= NB) return;
  const float* row = mc + b * LL;
  int   pick[NTOP];
  float pv[NTOP];
  for (int j = 0; j < NTOP; ++j) {
    float best = -3.4e38f; int bi = 0;
    for (int t = 0; t < LL; ++t) {
      bool skip = false;
      for (int p = 0; p < j; ++p) skip |= (pick[p] == t);
      float v = row[t];
      if (!skip && v > best) { best = v; bi = t; }
    }
    pick[j] = bi; pv[j] = best;
  }
  float m = pv[0], s = 0.f, e[NTOP];
  for (int j = 0; j < NTOP; ++j) { e[j] = expf(pv[j] - m); s += e[j]; }
  for (int j = 0; j < NTOP; ++j) { wts[b * NTOP + j] = e[j] / s; dly[b * NTOP + j] = pick[j]; }
}

// out[b,l,d] = sum_j w[b,j] * v[b,(l+delay_j)%L,d]
__global__ void agg_kernel(const float* __restrict__ v, const float* __restrict__ wts,
                           const int* __restrict__ dly, float* __restrict__ out) {
  long i = blockIdx.x * (long)blockDim.x + threadIdx.x;
  if (i >= BLD) return;
  int d = (int)(i % DDIM);
  long r = i / DDIM;
  int l = (int)(r % LL);
  int b = (int)(r / LL);
  float s = 0.f;
  for (int j = 0; j < NTOP; ++j) {
    int dl = dly[b * NTOP + j];
    s += wts[b * NTOP + j] * v[((long)b * LL + ((l + dl) & (LL - 1))) * DDIM + d];
  }
  out[i] = s;
}

// series_decomp: replicate-padded moving average k=25; optional trend accumulate
__global__ void decomp_kernel(const float* __restrict__ x, float* sea, float* trd,
                              int accum, int C, long total) {
  long i = blockIdx.x * (long)blockDim.x + threadIdx.x;
  if (i >= total) return;
  int c = (int)(i % C);
  long r = i / C;
  int l = (int)(r % LL);
  long base = (r - l) * (long)C + c;    // [b,0,c]
  float s = 0.f;
  for (int off = -12; off <= 12; ++off) {
    int ll = l + off;
    ll = ll < 0 ? 0 : (ll > LL - 1 ? LL - 1 : ll);
    s += x[base + (long)ll * C];
  }
  s *= (1.0f / 25.0f);
  if (sea) sea[i] = x[i] - s;
  if (trd) { if (accum) trd[i] += s; else trd[i] = s; }
}

// DataEmbedding_wo_pos: circular conv1d(k=3) over 7 ch + mark @ time
__global__ void embed_kernel(const float* __restrict__ x, const float* __restrict__ mark,
                             const float* __restrict__ token, const float* __restrict__ timew,
                             float* __restrict__ out) {
  long i = blockIdx.x * (long)blockDim.x + threadIdx.x;
  if (i >= BLD) return;
  int dm = (int)(i % DDIM);
  long r = i / DDIM;
  int l = (int)(r % LL);
  int b = (int)(r / LL);
  float s = 0.f;
  for (int kk = 0; kk < 3; ++kk) {
    int ll = (l + kk - 1 + LL) & (LL - 1);
    const float* xp = x + ((long)b * LL + ll) * CINC;
    for (int ci = 0; ci < CINC; ++ci)
      s += xp[ci] * token[(kk * CINC + ci) * DDIM + dm];
  }
  const float* mp = mark + ((long)b * LL + l) * DMK;
  for (int m = 0; m < DMK; ++m) s += mp[m] * timew[m * DDIM + dm];
  out[i] = s;
}

// LayerNorm over D=512 (one block per row)
__global__ void ln_kernel(const float* __restrict__ x, const float* __restrict__ w,
                          const float* __restrict__ b, float* __restrict__ out) {
  __shared__ float red[256];
  long row = blockIdx.x;
  const float* xr = x + row * DDIM;
  int t = threadIdx.x;
  float v0 = xr[t], v1 = xr[t + 256];
  red[t] = v0 + v1;
  __syncthreads();
  for (int s = 128; s > 0; s >>= 1) { if (t < s) red[t] += red[t + s]; __syncthreads(); }
  float mu = red[0] * (1.0f / 512.0f);
  __syncthreads();
  float d0 = v0 - mu, d1 = v1 - mu;
  red[t] = d0 * d0 + d1 * d1;
  __syncthreads();
  for (int s = 128; s > 0; s >>= 1) { if (t < s) red[t] += red[t + s]; __syncthreads(); }
  float rs = rsqrtf(red[0] * (1.0f / 512.0f) + 1e-5f);
  out[row * DDIM + t]       = d0 * rs * w[t] + b[t];
  out[row * DDIM + t + 256] = d1 * rs * w[t + 256] + b[t + 256];
}

__global__ void colmean_kernel(const float* __restrict__ x, float* __restrict__ cm) {
  int i = blockIdx.x * blockDim.x + threadIdx.x;
  if (i >= NB * DDIM) return;
  int d = i % DDIM, b = i / DDIM;
  float s = 0.f;
  for (int l = 0; l < LL; ++l) s += x[((long)b * LL + l) * DDIM + d];
  cm[i] = s * (1.0f / 512.0f);
}

__global__ void colsub_kernel(const float* __restrict__ x, const float* __restrict__ cm,
                              float* __restrict__ out) {
  long i = blockIdx.x * (long)blockDim.x + threadIdx.x;
  if (i >= BLD) return;
  int d = (int)(i % DDIM);
  int b = (int)((i / DDIM) / LL);
  out[i] = x[i] - cm[b * DDIM + d];
}

__global__ void mean_time_kernel(const float* __restrict__ x, float* __restrict__ out) {
  int i = blockIdx.x * blockDim.x + threadIdx.x;
  if (i >= NB * CINC) return;
  int c = i % CINC, b = i / CINC;
  float s = 0.f;
  for (int l = 0; l < LL; ++l) s += x[((long)b * LL + l) * CINC + c];
  out[i] = s * (1.0f / 512.0f);
}

__global__ void build_init_kernel(const float* __restrict__ tinit, const float* __restrict__ sinit,
                                  const float* __restrict__ mean7,
                                  float* __restrict__ tfull, float* __restrict__ sfull) {
  long i = blockIdx.x * (long)blockDim.x + threadIdx.x;
  if (i >= (long)NB * LL * CINC) return;
  int c = (int)(i % CINC);
  long r = i / CINC;
  int l = (int)(r % LL);
  int b = (int)(r / LL);
  if (l < 256) {
    long src = ((long)b * LL + (l + 256)) * CINC + c;
    tfull[i] = tinit[src];
    sfull[i] = sinit[src];
  } else {
    tfull[i] = mean7[b * CINC + c];
    sfull[i] = 0.f;
  }
}

__global__ void proj_kernel(const float* __restrict__ x, const float* __restrict__ pw,
                            const float* __restrict__ pb, float* __restrict__ out) {
  long i = blockIdx.x * (long)blockDim.x + threadIdx.x;
  if (i >= (long)NB * LL * CINC) return;
  int c = (int)(i % CINC);
  long row = i / CINC;
  const float* xr = x + row * DDIM;
  float s = pb[c];
  for (int k = 0; k < DDIM; ++k) s += xr[k] * pw[k * CINC + c];
  out[i] = s;
}

__global__ void trend_conv_kernel(const float* __restrict__ tsum, const float* __restrict__ tw,
                                  float* __restrict__ tacc) {
  long i = blockIdx.x * (long)blockDim.x + threadIdx.x;
  if (i >= (long)NB * LL * CINC) return;
  int co = (int)(i % CINC);
  long r = i / CINC;
  int l = (int)(r % LL);
  int b = (int)(r / LL);
  float s = 0.f;
  for (int kk = 0; kk < 3; ++kk) {
    int ll = (l + kk - 1 + LL) & (LL - 1);
    const float* tp = tsum + ((long)b * LL + ll) * DDIM;
    const float* wp = tw + (long)kk * DDIM * CINC + co;
    for (int ci = 0; ci < DDIM; ++ci) s += tp[ci] * wp[ci * CINC];
  }
  tacc[i] += s;
}

__global__ void final_kernel(const float* __restrict__ tf, const float* __restrict__ so,
                             float* __restrict__ out) {
  long i = blockIdx.x * (long)blockDim.x + threadIdx.x;
  if (i >= (long)NB * PREDN * CINC) return;
  int c = (int)(i % CINC);
  long r = i / CINC;
  int l = (int)(r % PREDN);
  int b = (int)(r / PREDN);
  long src = ((long)b * LL + (l + 256)) * CINC + c;
  out[i] = tf[src] + so[src];
}

__global__ void zero_kernel(float* __restrict__ p, long n) {
  long i = blockIdx.x * (long)blockDim.x + threadIdx.x;
  if (i < n) p[i] = 0.f;
}

// ---------------------------------------------------------------------------
// d_in flattening (setup_inputs leaf order):
//  0 x_enc  1 x_mark_enc  2 x_dec(unused)  3 x_mark_dec
//  4 enc_emb.token  5 enc_emb.time  6 dec_emb.token  7 dec_emb.time
//  enc layer0: 8 q_w 9 q_b 10 k_w 11 k_b 12 v_w 13 v_b 14 o_w 15 o_b 16 ff1 17 ff2
//  enc layer1: 18..27
//  28 enc_norm.w 29 enc_norm.b
//  dec layer0: 30..37 self, 38..45 cross, 46 ff1, 47 ff2, 48 trend_w
//  49 dec_norm.w 50 dec_norm.b 51 proj_w 52 proj_b
// ---------------------------------------------------------------------------
extern "C" void kernel_launch(void* const* d_in, const int* in_sizes, int n_in,
                              void* d_out, int out_size, void* d_ws, size_t ws_size,
                              hipStream_t stream) {
  (void)in_sizes; (void)n_in; (void)out_size; (void)ws_size;
  auto F = [&](int i) { return (const float*)d_in[i]; };
  const float* x_enc      = F(0);
  const float* x_mark_enc = F(1);
  const float* x_mark_dec = F(3);

  float* ws = (float*)d_ws;
  size_t off = 0;
  auto alloc = [&](long n) { float* p = ws + off; off += (size_t)n; return p; };

  float* buf_x   = alloc(BLD);
  float* buf_enc = alloc(BLD);
  float* buf_q   = alloc(BLD);
  float* buf_k   = alloc(BLD);
  float* buf_v   = alloc(BLD);
  float* buf_a   = alloc(BLD);
  float* buf_b   = alloc(BLD);
  float* buf_h   = alloc((long)NB * LL * DFF);
  float* buf_g   = alloc((long)NB * LL * LL);
  float* buf_ts  = alloc(BLD);
  float* mc      = alloc((long)NB * LL);
  float* wts     = alloc(NB * NTOP);
  int*   dly     = (int*)alloc(NB * NTOP);
  float* mean7   = alloc(NB * CINC);
  float* sinit   = alloc((long)NB * LL * CINC);
  float* tinit   = alloc((long)NB * LL * CINC);
  float* s_full  = alloc((long)NB * LL * CINC);
  float* t_acc   = alloc((long)NB * LL * CINC);
  float* s_out   = alloc((long)NB * LL * CINC);
  float* cm      = alloc(NB * DDIM);

  auto g1 = [&](long n) { return dim3((unsigned)((n + 255) / 256)); };

  auto run_gemm = [&](const float* A, const float* W, int transB, const float* bias,
                      const float* R, float* C, int M, int N, int K, int act,
                      int batch, long sA, long sW, long sR, long sC) {
    dim3 g((unsigned)(N / 64), (unsigned)(M / 128), (unsigned)batch);
    if (transB)
      gemm_wmma<1, 0><<<g, dim3(128), 0, stream>>>(A, sA, W, sW, bias, R, sR, C, sC, M, N, K);
    else if (act)
      gemm_wmma<0, 1><<<g, dim3(128), 0, stream>>>(A, sA, W, sW, bias, R, sR, C, sC, M, N, K);
    else
      gemm_wmma<0, 0><<<g, dim3(128), 0, stream>>>(A, sA, W, sW, bias, R, sR, C, sC, M, N, K);
  };

  auto run_attn = [&](const float* xq, const float* xkv, int wb, float* outb) {
    run_gemm(xq,  F(wb + 0), 0, F(wb + 1), nullptr, buf_q, (int)BL, DDIM, DDIM, 0, 1, 0, 0, 0, 0);
    run_gemm(xkv, F(wb + 2), 0, F(wb + 3), nullptr, buf_k, (int)BL, DDIM, DDIM, 0, 1, 0, 0, 0, 0);
    run_gemm(xkv, F(wb + 4), 0, F(wb + 5), nullptr, buf_v, (int)BL, DDIM, DDIM, 0, 1, 0, 0, 0, 0);
    // G[b] = Q[b] (L x D) . K[b]^T  -> per-batch WMMA GEMM (B operand transposed)
    run_gemm(buf_q, buf_k, 1, nullptr, nullptr, buf_g, LL, LL, DDIM, 0, NB,
             (long)LL * DDIM, (long)LL * DDIM, 0, (long)LL * LL);
    mean_corr_kernel<<<dim3(NB), dim3(256), 0, stream>>>(buf_g, mc);
    topk_kernel<<<dim3(1), dim3(32), 0, stream>>>(mc, wts, dly);
    agg_kernel<<<g1(BLD), dim3(256), 0, stream>>>(buf_v, wts, dly, buf_a);
    run_gemm(buf_a, F(wb + 6), 0, F(wb + 7), xq, outb, (int)BL, DDIM, DDIM, 0, 1, 0, 0, 0, 0);
  };

  auto my_ln = [&](const float* x, const float* w, const float* bb, float* tmp, float* outp) {
    ln_kernel<<<dim3((unsigned)BL), dim3(256), 0, stream>>>(x, w, bb, tmp);
    colmean_kernel<<<g1(NB * DDIM), dim3(256), 0, stream>>>(tmp, cm);
    colsub_kernel<<<g1(BLD), dim3(256), 0, stream>>>(tmp, cm, outp);
  };

  long nsm = (long)NB * LL * CINC;

  // --- init: mean, decomp(x_enc), build seasonal/trend init ---
  mean_time_kernel<<<g1(NB * CINC), dim3(256), 0, stream>>>(x_enc, mean7);
  decomp_kernel<<<g1(nsm), dim3(256), 0, stream>>>(x_enc, sinit, tinit, 0, CINC, nsm);
  build_init_kernel<<<g1(nsm), dim3(256), 0, stream>>>(tinit, sinit, mean7, t_acc, s_full);

  // --- encoder ---
  embed_kernel<<<g1(BLD), dim3(256), 0, stream>>>(x_enc, x_mark_enc, F(4), F(5), buf_x);
  const int encbase[2] = {8, 18};
  for (int li = 0; li < 2; ++li) {
    int wb = encbase[li];
    run_attn(buf_x, buf_x, wb, buf_b);
    decomp_kernel<<<g1(BLD), dim3(256), 0, stream>>>(buf_b, buf_x, (float*)nullptr, 0, DDIM, BLD);
    run_gemm(buf_x, F(wb + 8), 0, nullptr, nullptr, buf_h, (int)BL, DFF, DDIM, 1, 1, 0, 0, 0, 0);
    run_gemm(buf_h, F(wb + 9), 0, nullptr, buf_x, buf_b, (int)BL, DDIM, DFF, 0, 1, 0, 0, 0, 0);
    decomp_kernel<<<g1(BLD), dim3(256), 0, stream>>>(buf_b, buf_x, (float*)nullptr, 0, DDIM, BLD);
  }
  my_ln(buf_x, F(28), F(29), buf_a, buf_enc);

  // --- decoder ---
  embed_kernel<<<g1(BLD), dim3(256), 0, stream>>>(s_full, x_mark_dec, F(6), F(7), buf_x);
  zero_kernel<<<g1(BLD), dim3(256), 0, stream>>>(buf_ts, BLD);

  run_attn(buf_x, buf_x, 30, buf_b);                                   // self
  decomp_kernel<<<g1(BLD), dim3(256), 0, stream>>>(buf_b, buf_x, buf_ts, 1, DDIM, BLD);
  run_attn(buf_x, buf_enc, 38, buf_b);                                 // cross
  decomp_kernel<<<g1(BLD), dim3(256), 0, stream>>>(buf_b, buf_x, buf_ts, 1, DDIM, BLD);
  run_gemm(buf_x, F(46), 0, nullptr, nullptr, buf_h, (int)BL, DFF, DDIM, 1, 1, 0, 0, 0, 0);
  run_gemm(buf_h, F(47), 0, nullptr, buf_x, buf_b, (int)BL, DDIM, DFF, 0, 1, 0, 0, 0, 0);
  decomp_kernel<<<g1(BLD), dim3(256), 0, stream>>>(buf_b, buf_x, buf_ts, 1, DDIM, BLD);

  trend_conv_kernel<<<g1(nsm), dim3(256), 0, stream>>>(buf_ts, F(48), t_acc);

  my_ln(buf_x, F(49), F(50), buf_b, buf_a);
  proj_kernel<<<g1(nsm), dim3(256), 0, stream>>>(buf_a, F(51), F(52), s_out);

  final_kernel<<<g1((long)NB * PREDN * CINC), dim3(256), 0, stream>>>(t_acc, s_out, (float*)d_out);
}
